// MambaModel_37074157699730
// MI455X (gfx1250) — matrix operand order repaired and checked by reading, not compiled, not run on hardware
//
#include <hip/hip_runtime.h>

typedef __attribute__((ext_vector_type(16))) __bf16 v16bf;
typedef __attribute__((ext_vector_type(8)))  __bf16 v8bf;
typedef __attribute__((ext_vector_type(4)))  __bf16 v4bf;
typedef __attribute__((ext_vector_type(8)))  float  v8f;

#define L_SEQ   1024
#define DMODEL  768
#define DIN     1536
#define NSTATE  16
#define DTRANK  48
#define KCONV   4
#define VOCAB   1024
#define NLAYERS 8
#define DTK     64      // dt_proj K padded 48 -> 64 (multiple of 32 for WMMA)
#define XDBW    (DTRANK + 2*NSTATE)   // 80
#define EPSV    1e-5f

__device__ __forceinline__ float siluf(float x) { return x / (1.f + __expf(-x)); }

// ---------------- converters ----------------
__global__ void k_f32_to_bf16_v4(v4bf* __restrict__ dst, const float4* __restrict__ src, int n4) {
  int i = blockIdx.x * blockDim.x + threadIdx.x;
  if (i < n4) {
    float4 f = src[i];
    v4bf o;
    o[0] = (__bf16)f.x; o[1] = (__bf16)f.y; o[2] = (__bf16)f.z; o[3] = (__bf16)f.w;
    dst[i] = o;
  }
}

// dt_proj weight (NLAYERS*DIN, 48) -> (NLAYERS*DIN, 64) zero-padded bf16
__global__ void k_pad_dt_w(__bf16* __restrict__ dst, const float* __restrict__ src, int total) {
  int i = blockIdx.x * blockDim.x + threadIdx.x;
  if (i >= total) return;
  int k = i & (DTK - 1);
  int nd = i >> 6;
  dst[i] = (k < DTRANK) ? (__bf16)src[nd * DTRANK + k] : (__bf16)0.f;
}

// xdb[:, :48] -> (L, 64) zero-padded bf16 (A matrix for dt_proj GEMM)
__global__ void k_pack_dtin(__bf16* __restrict__ dst, const float* __restrict__ xdb, int total) {
  int i = blockIdx.x * blockDim.x + threadIdx.x;
  if (i >= total) return;
  int k = i & (DTK - 1);
  int m = i >> 6;
  dst[i] = (k < DTRANK) ? (__bf16)xdb[m * XDBW + k] : (__bf16)0.f;
}

// ---------------- embedding ----------------
__global__ void k_embed(float* __restrict__ x, const float* __restrict__ embed,
                        const int* __restrict__ src) {
  int l = blockIdx.x;
  int id = src[l];
  const float* e = embed + (long)id * DMODEL;
  for (int d = threadIdx.x; d < DMODEL; d += blockDim.x)
    x[(long)l * DMODEL + d] = e[d];
}

// ---------------- RMSNorm (row-per-block), bf16 out ----------------
__global__ __launch_bounds__(256)
void k_rmsnorm(__bf16* __restrict__ out, const float* __restrict__ x,
               const float* __restrict__ w) {
  int m = blockIdx.x;
  __shared__ float red[256];
  float s = 0.f;
  for (int d = threadIdx.x; d < DMODEL; d += 256) {
    float v = x[(long)m * DMODEL + d];
    s += v * v;
  }
  red[threadIdx.x] = s;
  __syncthreads();
  for (int o = 128; o > 0; o >>= 1) {
    if (threadIdx.x < o) red[threadIdx.x] += red[threadIdx.x + o];
    __syncthreads();
  }
  float scale = rsqrtf(red[0] / (float)DMODEL + EPSV);
  for (int d = threadIdx.x; d < DMODEL; d += 256)
    out[(long)m * DMODEL + d] = (__bf16)(x[(long)m * DMODEL + d] * scale * w[d]);
}

// ---------------- depthwise causal conv (K=4) + SiLU ----------------
__global__ __launch_bounds__(256)
void k_conv_silu(__bf16* __restrict__ u_bf, float* __restrict__ u_f,
                 const float* __restrict__ proj, const float* __restrict__ cw,
                 const float* __restrict__ cb) {
  int d = blockIdx.x * 256 + threadIdx.x;   // channel
  int l = blockIdx.y;                       // time
  float acc = cb[d];
#pragma unroll
  for (int j = 0; j < KCONV; ++j) {
    int t = l - (KCONV - 1) + j;
    if (t >= 0) acc += proj[(long)t * (2 * DIN) + d] * cw[d * KCONV + j];
  }
  acc = siluf(acc);
  u_f[(long)l * DIN + d] = acc;
  u_bf[(long)l * DIN + d] = (__bf16)acc;
}

// ---------------- bf16 WMMA GEMM: C[m][n] = sum_k A[m][k]*W[n][k] ----------------
// 4-wave block, 64x64 tile, K stepped by 32, double-buffered LDS.
// A tile: async global->LDS DMA (ASYNCcnt), issued one step ahead of compute.
// W tile: b128 vector loads (compile-time row guard via BOUND).
// MODE 0: store ; MODE 1: softplus(acc + bias[n]) ; MODE 2: C += acc (residual)
#define TM 64
#define TN 64
#define TKK 32
#define LDT 40   // LDS row pitch in bf16: 80 B, 16B-aligned rows, staggers banks
#define ABUF (TM * LDT * 2)   // bytes per A buffer (== per W buffer)

template<int MODE, bool BOUND>
__global__ __launch_bounds__(128)
void k_gemm_bf16(const __bf16* __restrict__ A, const __bf16* __restrict__ W,
                 float* __restrict__ C, int N, int K, int ldc,
                 const float* __restrict__ bias) {
  __shared__ alignas(16) __bf16 As[2][TM][LDT];
  __shared__ alignas(16) __bf16 Ws[2][TN][LDT];
  const int tid  = threadIdx.x;
  const int wv   = tid >> 5;
  const int lane = tid & 31;
  const int lm   = lane & 15;
  const int g    = lane >> 4;
  const int n0   = blockIdx.x * TN;
  const int m0   = blockIdx.y * TM;

  // hoisted per-thread tile-mover state: rows r and r+32, one 16-B chunk each
  const int r  = tid >> 2;
  const int cb = (tid & 3) << 4;
  unsigned long long gaA0 =
      (unsigned long long)(uintptr_t)A + ((long)(m0 + r) * K) * 2 + cb;
  unsigned long long gaA1 =
      (unsigned long long)(uintptr_t)A + ((long)(m0 + r + 32) * K) * 2 + cb;
  const char* wp0 = (const char*)W + ((long)(n0 + r) * K) * 2 + cb;
  const char* wp1 = (const char*)W + ((long)(n0 + r + 32) * K) * 2 + cb;
  const bool valid0 = (n0 + r) < N;
  const bool valid1 = (n0 + r + 32) < N;
  const unsigned lofA0 = (unsigned)(uintptr_t)(&As[0][r][0]) + (unsigned)cb;
  const unsigned lofA1 = (unsigned)(uintptr_t)(&As[0][r + 32][0]) + (unsigned)cb;
  char* wsb0 = (char*)&Ws[0][r][0] + cb;
  char* wsb1 = (char*)&Ws[0][r + 32][0] + cb;

  auto issue = [&](int b) {
    unsigned off = (unsigned)b * ABUF;
    asm volatile("global_load_async_to_lds_b128 %0, %1, off"
                 :: "v"(lofA0 + off), "v"(gaA0) : "memory");
    asm volatile("global_load_async_to_lds_b128 %0, %1, off"
                 :: "v"(lofA1 + off), "v"(gaA1) : "memory");
    gaA0 += 2 * TKK; gaA1 += 2 * TKK;
    uint4 w0{}, w1{};
    if (!BOUND || valid0) w0 = *(const uint4*)wp0;
    if (!BOUND || valid1) w1 = *(const uint4*)wp1;
    wp0 += 2 * TKK; wp1 += 2 * TKK;
    *(uint4*)(wsb0 + off) = w0;
    *(uint4*)(wsb1 + off) = w1;
  };

  v8f acc[4] = {};
  const int arow = 16 * wv + lm;
  const int nsteps = K / TKK;

  issue(0);
  for (int s = 0; s < nsteps; ++s) {
    const int cur = s & 1;
    asm volatile("s_wait_asynccnt 0x0" ::: "memory");  // tile `cur` DMA complete
    __syncthreads();                                   // + all waves' W stores visible
    if (s + 1 < nsteps) issue(cur ^ 1);                // DMA next tile under compute

    // A fragment: lane -> row (16*wv + lm); K: [g*8 .. g*8+7] and [16+g*8 ..]
    v8bf a0 = *(const v8bf*)&As[cur][arow][g * 8];
    v8bf a1 = *(const v8bf*)&As[cur][arow][16 + g * 8];
    v16bf a = __builtin_shufflevector(a0, a1, 0, 1, 2, 3, 4, 5, 6, 7,
                                      8, 9, 10, 11, 12, 13, 14, 15);
#pragma unroll
    for (int j = 0; j < 4; ++j) {
      // B fragment: lane -> col (16*j + lm); K: [g*8 .. g*8+7] and [16+g*8 ..]; B[k][n] = W[n][k]
      v8bf b0 = *(const v8bf*)&Ws[cur][16 * j + lm][g * 8];
      v8bf b1 = *(const v8bf*)&Ws[cur][16 * j + lm][16 + g * 8];
      v16bf b = __builtin_shufflevector(b0, b1, 0, 1, 2, 3, 4, 5, 6, 7,
                                        8, 9, 10, 11, 12, 13, 14, 15);
      acc[j] = __builtin_amdgcn_wmma_f32_16x16x32_bf16(false, a, false, b,
                                                       (short)0, acc[j], false, false);
    }
    __syncthreads();   // all reads of buffer `cur` done before it is overwritten
  }

  // epilogue: C/D layout -> lane col n = lm, VGPR v -> row v + 8*g
#pragma unroll
  for (int j = 0; j < 4; ++j) {
    int n = n0 + 16 * j + lm;
    if (!BOUND || n < N) {
#pragma unroll
      for (int v = 0; v < 8; ++v) {
        int m = m0 + 16 * wv + v + 8 * g;
        float val = acc[j][v];
        if (MODE == 1) {
          val += bias[n];
          val = (val > 20.f) ? val : log1pf(__expf(val));
        }
        if (MODE == 2) C[(long)m * ldc + n] += val;
        else           C[(long)m * ldc + n] = val;
      }
    }
  }
}

// ---------------- selective scan ----------------
// Thread per (channel, state): 16 channels x 16 states per 256-thread block,
// 96 blocks cover DIN=1536. One v_exp_f32 per step per thread; y reduced over
// the 16 state lanes with __shfl_xor (stays inside a wave32 half). dt/u/gate/B/C
// staged per 64-step chunk in LDS; y written back coalesced (gate+D fused).
__global__ __launch_bounds__(256)
void k_scan(const float* __restrict__ dt, const float* __restrict__ xdb,
            const float* __restrict__ u, const float* __restrict__ proj,
            const float* __restrict__ alog, const float* __restrict__ dpar,
            __bf16* __restrict__ y_bf) {
  __shared__ float Bsh[64][NSTATE];
  __shared__ float Csh[64][NSTATE];
  __shared__ float dtsh[64][16];
  __shared__ float ush[64][16];
  __shared__ float gsh[64][16];
  __shared__ float ysh[64][16];
  __shared__ float Dpsh[16];

  const int tid  = threadIdx.x;
  const int n    = tid & 15;          // state index
  const int dloc = tid >> 4;          // local channel 0..15
  const int d0   = blockIdx.x * 16;
  const int d    = d0 + dloc;

  if (tid < 16) Dpsh[tid] = dpar[d0 + tid];
  float Aval = -__expf(alog[(long)d * NSTATE + n]);
  float h = 0.f;

  for (int t0 = 0; t0 < L_SEQ; t0 += 64) {
    // stage chunk: each thread moves one float4 per array
    {
      int r  = tid >> 2;
      int c4 = (tid & 3) << 2;
      *(float4*)&Bsh[r][c4]  = *(const float4*)&xdb[(long)(t0 + r) * XDBW + DTRANK + c4];
      *(float4*)&Csh[r][c4]  = *(const float4*)&xdb[(long)(t0 + r) * XDBW + DTRANK + 16 + c4];
      *(float4*)&dtsh[r][c4] = *(const float4*)&dt[(long)(t0 + r) * DIN + d0 + c4];
      *(float4*)&ush[r][c4]  = *(const float4*)&u[(long)(t0 + r) * DIN + d0 + c4];
      *(float4*)&gsh[r][c4]  = *(const float4*)&proj[(long)(t0 + r) * (2 * DIN) + DIN + d0 + c4];
    }
    __syncthreads();

    for (int tt = 0; tt < 64; ++tt) {
      float dtv = dtsh[tt][dloc];               // LDS broadcast
      float uv  = ush[tt][dloc];
      float dA  = __expf(dtv * Aval);
      h = dA * h + (dtv * uv) * Bsh[tt][n];
      float yv = h * Csh[tt][n];
      yv += __shfl_xor(yv, 1);
      yv += __shfl_xor(yv, 2);
      yv += __shfl_xor(yv, 4);
      yv += __shfl_xor(yv, 8);
      if (n == 0) ysh[tt][dloc] = yv;
    }
    __syncthreads();

    // coalesced writeback with D skip-connection and SiLU gate fused
    for (int i = tid; i < 64 * 16; i += 256) {
      int rr = i >> 4, c = i & 15;
      float val = (ysh[rr][c] + ush[rr][c] * Dpsh[c]) * siluf(gsh[rr][c]);
      y_bf[(long)(t0 + rr) * DIN + d0 + c] = (__bf16)val;
    }
    __syncthreads();
  }
}

// ---------------- host launcher ----------------
extern "C" void kernel_launch(void* const* d_in, const int* in_sizes, int n_in,
                              void* d_out, int out_size, void* d_ws, size_t ws_size,
                              hipStream_t stream) {
  const int*   src    = (const int*)  d_in[0];
  const float* embed  = (const float*)d_in[1];
  const float* norm_w = (const float*)d_in[2];
  const float* in_w   = (const float*)d_in[3];
  const float* conv_w = (const float*)d_in[4];
  const float* conv_b = (const float*)d_in[5];
  const float* x_w    = (const float*)d_in[6];
  const float* dt_w   = (const float*)d_in[7];
  const float* dt_b   = (const float*)d_in[8];
  const float* alog   = (const float*)d_in[9];
  const float* dpar   = (const float*)d_in[10];
  const float* out_w  = (const float*)d_in[11];
  const float* fnw    = (const float*)d_in[12];
  const float* lm_w   = (const float*)d_in[13];

  char* p = (char*)d_ws;
  auto alloc = [&](size_t bytes) -> void* {
    void* r = (void*)p;
    p += (bytes + 255) & ~(size_t)255;
    return r;
  };
  __bf16* w_in_bf  = (__bf16*)alloc((size_t)NLAYERS * 2 * DIN * DMODEL * 2);
  __bf16* w_x_bf   = (__bf16*)alloc((size_t)NLAYERS * XDBW * DIN * 2);
  __bf16* w_dt_bf  = (__bf16*)alloc((size_t)NLAYERS * DIN * DTK * 2);
  __bf16* w_out_bf = (__bf16*)alloc((size_t)NLAYERS * DMODEL * DIN * 2);
  __bf16* w_lm_bf  = (__bf16*)alloc((size_t)VOCAB * DMODEL * 2);
  float*  x        = (float*) alloc((size_t)L_SEQ * DMODEL * 4);
  __bf16* hn_bf    = (__bf16*)alloc((size_t)L_SEQ * DMODEL * 2);
  float*  proj     = (float*) alloc((size_t)L_SEQ * 2 * DIN * 4);
  __bf16* u_bf     = (__bf16*)alloc((size_t)L_SEQ * DIN * 2);
  float*  u_f      = (float*) alloc((size_t)L_SEQ * DIN * 4);
  float*  xdb      = (float*) alloc((size_t)L_SEQ * XDBW * 4);
  __bf16* dtin_bf  = (__bf16*)alloc((size_t)L_SEQ * DTK * 2);
  float*  dtv      = (float*) alloc((size_t)L_SEQ * DIN * 4);
  __bf16* y_bf     = (__bf16*)alloc((size_t)L_SEQ * DIN * 2);

  auto cvt = [&](__bf16* dst, const float* s, size_t n) {
    int n4 = (int)(n / 4);
    k_f32_to_bf16_v4<<<dim3((n4 + 255) / 256), dim3(256), 0, stream>>>(
        (v4bf*)dst, (const float4*)s, n4);
  };
  cvt(w_in_bf,  in_w,  (size_t)NLAYERS * 2 * DIN * DMODEL);
  cvt(w_x_bf,   x_w,   (size_t)NLAYERS * XDBW * DIN);
  cvt(w_out_bf, out_w, (size_t)NLAYERS * DMODEL * DIN);
  cvt(w_lm_bf,  lm_w,  (size_t)VOCAB * DMODEL);
  {
    int n = NLAYERS * DIN * DTK;
    k_pad_dt_w<<<dim3((n + 255) / 256), 256, 0, stream>>>(w_dt_bf, dt_w, n);
  }

  k_embed<<<dim3(L_SEQ), 256, 0, stream>>>(x, embed, src);

  for (int lay = 0; lay < NLAYERS; ++lay) {
    k_rmsnorm<<<dim3(L_SEQ), 256, 0, stream>>>(hn_bf, x, norm_w + (size_t)lay * DMODEL);

    k_gemm_bf16<0, false><<<dim3((2 * DIN) / TN, L_SEQ / TM), 128, 0, stream>>>(
        hn_bf, w_in_bf + (size_t)lay * 2 * DIN * DMODEL, proj,
        2 * DIN, DMODEL, 2 * DIN, nullptr);

    k_conv_silu<<<dim3(DIN / 256, L_SEQ), 256, 0, stream>>>(
        u_bf, u_f, proj, conv_w + (size_t)lay * DIN * KCONV, conv_b + (size_t)lay * DIN);

    k_gemm_bf16<0, true><<<dim3((XDBW + TN - 1) / TN, L_SEQ / TM), 128, 0, stream>>>(
        u_bf, w_x_bf + (size_t)lay * XDBW * DIN, xdb,
        XDBW, DIN, XDBW, nullptr);

    {
      int n = L_SEQ * DTK;
      k_pack_dtin<<<dim3((n + 255) / 256), 256, 0, stream>>>(dtin_bf, xdb, n);
    }

    k_gemm_bf16<1, false><<<dim3(DIN / TN, L_SEQ / TM), 128, 0, stream>>>(
        dtin_bf, w_dt_bf + (size_t)lay * DIN * DTK, dtv,
        DIN, DTK, DIN, dt_b + (size_t)lay * DIN);

    k_scan<<<dim3(DIN / 16), 256, 0, stream>>>(
        dtv, xdb, u_f, proj,
        alog + (size_t)lay * DIN * NSTATE, dpar + (size_t)lay * DIN, y_bf);

    k_gemm_bf16<2, false><<<dim3(DMODEL / TN, L_SEQ / TM), 128, 0, stream>>>(
        y_bf, w_out_bf + (size_t)lay * DMODEL * DIN, x,
        DMODEL, DIN, DMODEL, nullptr);
  }

  k_rmsnorm<<<dim3(L_SEQ), 256, 0, stream>>>(hn_bf, x, fnw);
  k_gemm_bf16<0, false><<<dim3(VOCAB / TN, L_SEQ / TM), 128, 0, stream>>>(
      hn_bf, w_lm_bf, (float*)d_out, VOCAB, DMODEL, VOCAB, nullptr);
}
